// AtlasBlock_85126251806878
// MI455X (gfx1250) — matrix operand order repaired and checked by reading, not compiled, not run on hardware
//
#include <hip/hip_runtime.h>
#include <math.h>

// ---------------------------------------------------------------------------
// Types for CDNA5 WMMA (wave32): A/B = 16 halves per lane, C/D = 8 floats.
// ---------------------------------------------------------------------------
typedef _Float16 h16;
typedef h16   v16h __attribute__((ext_vector_type(16)));
typedef h16   h2   __attribute__((ext_vector_type(2)));
typedef float v8f  __attribute__((ext_vector_type(8)));
typedef float f2   __attribute__((ext_vector_type(2)));

#define BM 128
#define BN 64
#define BK 32
#define PA 34        // As row pitch in floats (BK + 2) -> 8B-aligned pairs
#define PB 66        // Bs row pitch in floats (BN + 2)

enum { EPI_NONE = 0, EPI_GELU = 1, EPI_SIGMOID = 2 };

__device__ __forceinline__ float gelu_tanh(float v) {
    const float c = 0.7978845608028654f;
    float u = c * (v + 0.044715f * v * v * v);
    return 0.5f * v * (1.0f + tanhf(u));
}

// byte offset of a generic pointer inside LDS (addrspace(3) view)
__device__ __forceinline__ unsigned lds_off(const void* p) {
    return (unsigned)(unsigned long long)
        (const __attribute__((address_space(3))) void*)p;
}

// async DMA: 16B / 4B per lane, global -> LDS, tracked by ASYNCcnt
__device__ __forceinline__ void async_b128(unsigned lds, const float* g) {
    asm volatile("global_load_async_to_lds_b128 %0, %1, off"
                 :: "v"(lds), "v"((unsigned long long)g) : "memory");
}
__device__ __forceinline__ void async_b32(unsigned lds, const float* g) {
    asm volatile("global_load_async_to_lds_b32 %0, %1, off"
                 :: "v"(lds), "v"((unsigned long long)g) : "memory");
}
__device__ __forceinline__ void wait_async0() {
    asm volatile("s_wait_asynccnt 0" ::: "memory");
}

// ---------------------------------------------------------------------------
// Generic batched GEMM:  C = epi( alpha*(A@B) + bias + rscale*Res )
//   A: MxK (row major, lda) or transA -> A[m,k]=Ap[k*lda+m]
//   B: KxN (row major, ldb) or transB -> B[k,n]=Bp[n*ldb+k]
//   grid.z decomposed as z -> (b, h, nb); per-operand affine offsets.
// Tiles: 128x64 block, 4 waves, 64x32 per wave (8 WMMA / k-step).
// Staging: double-buffered fp32 LDS filled by global_load_async_to_lds,
//          f32->f16 packing at fragment gather (v_cvt_pk_f16_f32).
// Requires: M%128==0, N%64==0, K%32==0 (true for every call in this model).
// ---------------------------------------------------------------------------
__global__ __launch_bounds__(128)
void gemm_wmma_f16(const float* __restrict__ A, const float* __restrict__ B,
                   float* __restrict__ C,
                   const float* __restrict__ bias,
                   const float* __restrict__ Res,
                   float alpha, float rscale,
                   int M, int N, int K,
                   long lda, long ldb, long ldc, long ldr,
                   int transA, int transB,
                   int dimH, int dimNB,
                   long sAb, long sAh, long sAn,
                   long sBb, long sBh, long sBn,
                   long sCb, long sCh, long sCn,
                   long sRb, int epilogue)
{
    __shared__ float As[2][BM * PA];   // [m][k]
    __shared__ float Bs[2][BK * PB];   // [k][n]

    // batch decomposition
    int z  = blockIdx.z;
    int nb = z % dimNB;
    int t  = z / dimNB;
    int h  = t % dimH;
    int b  = t / dimH;
    A += (long)b * sAb + (long)h * sAh + (long)nb * sAn;
    B += (long)b * sBb + (long)h * sBh + (long)nb * sBn;
    C += (long)b * sCb + (long)h * sCh + (long)nb * sCn;
    if (Res) Res += (long)b * sRb;

    const int tid  = threadIdx.x;
    const int lane = tid & 31;
    const int wave = tid >> 5;          // 0..3
    const int wm   = wave >> 1;         // wave row (0/1) -> 64 rows each
    const int wn   = wave & 1;          // wave col (0/1) -> 32 cols each
    const int m0   = blockIdx.y * BM;
    const int n0   = blockIdx.x * BN;
    const int lh   = lane & 15;
    const int hi   = lane >> 4;         // lane half (0/1)

    const unsigned asB[2] = { lds_off(&As[0][0]), lds_off(&As[1][0]) };
    const unsigned bsB[2] = { lds_off(&Bs[0][0]), lds_off(&Bs[1][0]) };

    // stage one (BMxBK, BKxBN) tile pair into buffer `buf` via async DMA
    auto stage = [&](int kt, int buf) {
        if (!transA) {                       // contiguous in k: 16B chunks
            for (int c = tid; c < BM * (BK / 4); c += 128) {
                int m = c >> 3, kq = (c & 7) << 2;
                async_b128(asB[buf] + (unsigned)(m * PA + kq) * 4u,
                           A + (long)(m0 + m) * lda + (kt + kq));
            }
        } else {                             // contiguous in m: 4B scatter
            for (int c = tid; c < BM * BK; c += 128) {
                int k = c >> 7, m = c & 127;
                async_b32(asB[buf] + (unsigned)(m * PA + k) * 4u,
                          A + (long)(kt + k) * lda + (m0 + m));
            }
        }
        if (!transB) {                       // contiguous in n: 16B chunks
            for (int c = tid; c < BK * (BN / 4); c += 128) {
                int k = c >> 4, nq = (c & 15) << 2;
                async_b128(bsB[buf] + (unsigned)(k * PB + nq) * 4u,
                           B + (long)(kt + k) * ldb + (n0 + nq));
            }
        } else {                             // contiguous in k: 4B scatter
            for (int c = tid; c < BK * BN; c += 128) {
                int n = c >> 5, k = c & 31;
                async_b32(bsB[buf] + (unsigned)(k * PB + n) * 4u,
                          B + (long)(n0 + n) * ldb + (kt + k));
            }
        }
    };

    const v8f vzero = {0.f,0.f,0.f,0.f,0.f,0.f,0.f,0.f};
    v8f acc[4][2];
    #pragma unroll
    for (int mi = 0; mi < 4; mi++) { acc[mi][0] = vzero; acc[mi][1] = vzero; }

    const int nsteps = K / BK;
    stage(0, 0);

    for (int s = 0; s < nsteps; s++) {
        const int buf = s & 1;
        wait_async0();                       // tile `s` resident in LDS
        __syncthreads();                     // ..visible to all waves; prior
                                             //   readers of buf^1 are done
        if (s + 1 < nsteps) stage((s + 1) * BK, buf ^ 1);

        const float* Ab = &As[buf][0];
        const float* Bb = &Bs[buf][0];

        // ---- gather WMMA fragments (ISA 7.12.2 lane maps), f32->f16 ----
        v16h af[4], bf[2];
        #pragma unroll
        for (int mi = 0; mi < 4; mi++) {
            int m = wm * 64 + mi * 16 + lh;
            union { v16h v; h2 p[8]; } u;
            #pragma unroll
            for (int p = 0; p < 8; p++) {
                int k0 = ((p < 4) ? 0 : 16) + hi * 8 + (p & 3) * 2; // A 16x32
                f2 w = *(const f2*)&Ab[m * PA + k0];
                h2 t; t.x = (h16)w.x; t.y = (h16)w.y;
                u.p[p] = t;
            }
            af[mi] = u.v;
        }
        #pragma unroll
        for (int ni = 0; ni < 2; ni++) {
            int n = wn * 32 + ni * 16 + lh;
            union { v16h v; h2 p[8]; } u;
            #pragma unroll
            for (int p = 0; p < 8; p++) {
                int k = ((p < 4) ? 0 : 16) + hi * 8 + (p & 3) * 2;  // B 32x16
                h2 t;
                t.x = (h16)Bb[k * PB + n];
                t.y = (h16)Bb[(k + 1) * PB + n];
                u.p[p] = t;
            }
            bf[ni] = u.v;
        }

        #pragma unroll
        for (int mi = 0; mi < 4; mi++)
            #pragma unroll
            for (int ni = 0; ni < 2; ni++)
                acc[mi][ni] = __builtin_amdgcn_wmma_f32_16x16x32_f16(
                    false, af[mi], false, bf[ni], (short)0, acc[mi][ni],
                    false, false);
        __syncthreads();                     // done reading buf
    }

    // ---- epilogue + store (C layout: lane=n, VGPR r -> m = r + hi*8) ----
    #pragma unroll
    for (int mi = 0; mi < 4; mi++) {
        #pragma unroll
        for (int ni = 0; ni < 2; ni++) {
            int nn = n0 + wn * 32 + ni * 16 + lh;
            float bv = bias ? bias[nn] : 0.0f;
            #pragma unroll
            for (int r = 0; r < 8; r++) {
                int mm = m0 + wm * 64 + mi * 16 + hi * 8 + r;
                float v = alpha * acc[mi][ni][r] + bv;
                if (Res) v += rscale * Res[(long)mm * ldr + nn];
                if (epilogue == EPI_GELU)         v = gelu_tanh(v);
                else if (epilogue == EPI_SIGMOID) v = 1.0f / (1.0f + __expf(-v));
                C[(long)mm * ldc + nn] = v;
            }
        }
    }
}

// ---------------------------------------------------------------------------
// LayerNorm: one block per row of D=1024.
// ---------------------------------------------------------------------------
__global__ __launch_bounds__(256)
void layernorm_kernel(const float* __restrict__ x, const float* __restrict__ g,
                      const float* __restrict__ bb, float* __restrict__ out, int D)
{
    long row = blockIdx.x;
    const float* xr = x + row * (long)D;
    float* orow = out + row * (long)D;
    __shared__ float red[256];

    float s = 0.f;
    for (int i = threadIdx.x; i < D; i += 256) s += xr[i];
    red[threadIdx.x] = s; __syncthreads();
    for (int st = 128; st > 0; st >>= 1) {
        if (threadIdx.x < st) red[threadIdx.x] += red[threadIdx.x + st];
        __syncthreads();
    }
    float mu = red[0] / D;
    __syncthreads();

    float vs = 0.f;
    for (int i = threadIdx.x; i < D; i += 256) { float d = xr[i] - mu; vs += d * d; }
    red[threadIdx.x] = vs; __syncthreads();
    for (int st = 128; st > 0; st >>= 1) {
        if (threadIdx.x < st) red[threadIdx.x] += red[threadIdx.x + st];
        __syncthreads();
    }
    float rstd = rsqrtf(red[0] / D + 1e-5f);
    for (int i = threadIdx.x; i < D; i += 256)
        orow[i] = (xr[i] - mu) * rstd * g[i] + bb[i];
}

// ---------------------------------------------------------------------------
// Banded-causal softmax over 2W=1024 keys, in place. rows = B*H*NB*W.
// mask(j) = (j > i) && (j <= W+i) && ((nb-1)*W + j >= 0); else score=-1e9.
// ---------------------------------------------------------------------------
__global__ __launch_bounds__(256)
void softmax_band(float* __restrict__ S, int NBc)
{
    const int W = 512, TW = 1024;
    long row = blockIdx.x;
    int  i   = (int)(row % W);
    long z   = row / W;
    int  nb  = (int)(z % NBc);
    float* sr = S + row * (long)TW;
    __shared__ float red[256];

    float vals[4];
    float mx = -1e30f;
    #pragma unroll
    for (int t = 0; t < 4; t++) {
        int j = threadIdx.x + t * 256;
        float v = sr[j];
        bool ok = (j > i) && (j <= W + i) && ((nb - 1) * W + j >= 0);
        v = ok ? v : -1e9f;
        vals[t] = v;
        mx = fmaxf(mx, v);
    }
    red[threadIdx.x] = mx; __syncthreads();
    for (int st = 128; st > 0; st >>= 1) {
        if (threadIdx.x < st) red[threadIdx.x] = fmaxf(red[threadIdx.x], red[threadIdx.x + st]);
        __syncthreads();
    }
    mx = red[0]; __syncthreads();

    float sum = 0.f;
    #pragma unroll
    for (int t = 0; t < 4; t++) { vals[t] = __expf(vals[t] - mx); sum += vals[t]; }
    red[threadIdx.x] = sum; __syncthreads();
    for (int st = 128; st > 0; st >>= 1) {
        if (threadIdx.x < st) red[threadIdx.x] += red[threadIdx.x + st];
        __syncthreads();
    }
    float inv = 1.0f / red[0];
    #pragma unroll
    for (int t = 0; t < 4; t++) sr[threadIdx.x + t * 256] = vals[t] * inv;
}

// ---------------------------------------------------------------------------
// ret_grad = ret_local*(W_temp - W_prev) + ret_global*W_temp,
// W_temp = W_prev - lr*(beta*m_prev + grad_full)
// ---------------------------------------------------------------------------
__global__ __launch_bounds__(256)
void retgrad_kernel(const float* __restrict__ Wp, const float* __restrict__ mp,
                    const float* __restrict__ g,
                    const float* __restrict__ lr_p, const float* __restrict__ rl_p,
                    const float* __restrict__ rg_p,
                    float* __restrict__ ret, long n)
{
    long i = (long)blockIdx.x * 256 + threadIdx.x;
    if (i >= n) return;
    float lr = *lr_p, rl = *rl_p, rg = *rg_p;
    float step = lr * (0.9f * mp[i] + g[i]);
    float Wt = Wp[i] - step;
    ret[i] = rl * (Wt - Wp[i]) + rg * Wt;
}

// mm = beta*mm + g ; Wm -= lr*(mm + ret_grad)
__global__ __launch_bounds__(256)
void memstep_kernel(float* __restrict__ Wm, float* __restrict__ mm,
                    const float* __restrict__ g, const float* __restrict__ ret,
                    const float* __restrict__ lr_p, long n)
{
    long i = (long)blockIdx.x * 256 + threadIdx.x;
    if (i >= n) return;
    float lr = *lr_p;
    float m2 = 0.9f * mm[i] + g[i];
    mm[i] = m2;
    Wm[i] = Wm[i] - lr * (m2 + ret[i]);
}

// x2 = x + gate*mem + (1-gate)*attn
__global__ __launch_bounds__(256)
void fuse_kernel(const float* __restrict__ x, const float* __restrict__ gate,
                 const float* __restrict__ mem, const float* __restrict__ attn,
                 float* __restrict__ x2, long n)
{
    long i = (long)blockIdx.x * 256 + threadIdx.x;
    if (i >= n) return;
    float gv = gate[i];
    x2[i] = x[i] + gv * mem[i] + (1.0f - gv) * attn[i];
}

// ---------------------------------------------------------------------------
extern "C" void kernel_launch(void* const* d_in, const int* in_sizes, int n_in,
                              void* d_out, int out_size, void* d_ws, size_t ws_size,
                              hipStream_t stream) {
    (void)in_sizes; (void)n_in; (void)out_size; (void)ws_size;

    const int  Bb = 2, S = 4096, D = 1024, H = 16, WIN = 512, CH = 512, HD = 64,
               NB = 8, NC = 8;
    const long BSD = (long)Bb * S * D;          // 8,388,608
    const long DD  = (long)D * D;               // 1,048,576
    const long BDD = (long)Bb * DD;
    const long SP  = (long)(S + WIN) * D;       // padded seq stride per batch
    const long SCB = (long)H * NB * WIN * 2 * WIN; // scores per-batch stride

    const float* x      = (const float*)d_in[0];
    const float* W_prev = (const float*)d_in[1];
    const float* m_prev = (const float*)d_in[2];
    const float* ln1_g  = (const float*)d_in[3];
    const float* ln1_b  = (const float*)d_in[4];
    const float* ln2_g  = (const float*)d_in[5];
    const float* ln2_b  = (const float*)d_in[6];
    const float* ln3_g  = (const float*)d_in[7];
    const float* ln3_b  = (const float*)d_in[8];
    const float* mem_Wq = (const float*)d_in[9];
    const float* mem_Wk = (const float*)d_in[10];
    const float* mem_Wv = (const float*)d_in[11];
    const float* mem_lr = (const float*)d_in[12];
    const float* ret_l  = (const float*)d_in[13];
    const float* ret_g  = (const float*)d_in[14];
    const float* aWq    = (const float*)d_in[15];
    const float* abq    = (const float*)d_in[16];
    const float* aWk    = (const float*)d_in[17];
    const float* abk    = (const float*)d_in[18];
    const float* aWv    = (const float*)d_in[19];
    const float* abv    = (const float*)d_in[20];
    const float* aWo    = (const float*)d_in[21];
    const float* abo    = (const float*)d_in[22];
    const float* gateW  = (const float*)d_in[23];
    const float* gateb  = (const float*)d_in[24];
    const float* fW1    = (const float*)d_in[25];
    const float* fb1    = (const float*)d_in[26];
    const float* fW2    = (const float*)d_in[27];
    const float* fb2    = (const float*)d_in[28];

    float* out_x = (float*)d_out;
    float* W_new = out_x + BSD;   // running Wm, becomes output in place
    float* m_new = W_new + BDD;   // running mm, becomes output in place

    // workspace carve-up
    float* ws = (float*)d_ws;
    float* xn      = ws;            ws += BSD;   // reused: xn1 / xn2 / xn3
    float* qm      = ws;            ws += BSD;
    float* km      = ws;            ws += BSD;
    float* vm      = ws;            ws += BSD;
    float* errbuf  = ws;            ws += BSD;   // full-seq err; chunk err reuses prefix
    float* mem_out = ws;            ws += BSD;
    float* qa      = ws;            ws += BSD;
    float* attn    = ws;            ws += BSD;
    float* attn_o  = ws;            ws += BSD;
    float* gate    = ws;            ws += BSD;
    float* x2      = ws;            ws += BSD;
    float* ka_pad  = ws;            ws += (long)Bb * SP;
    float* va_pad  = ws;            ws += (long)Bb * SP;
    float* gradb   = ws;            ws += BDD;
    float* retgr   = ws;            ws += BDD;
    float* hbuf    = ws;            ws += (long)Bb * S * 4 * D;
    float* scores  = ws;            ws += (long)Bb * SCB;

    auto gemm = [&](const float* A, const float* B, float* C, const float* bias,
                    const float* Res, float alpha, float rscale,
                    int M, int N, int K, long lda, long ldb, long ldc, long ldr,
                    int tA, int tB, int Z, int dimH, int dimNB,
                    long sAb, long sAh, long sAn,
                    long sBb, long sBh, long sBn,
                    long sCb, long sCh, long sCn,
                    long sRb, int epi) {
        dim3 grid(N / BN, M / BM, Z);
        gemm_wmma_f16<<<grid, dim3(128), 0, stream>>>(
            A, B, C, bias, Res, alpha, rscale, M, N, K, lda, ldb, ldc, ldr,
            tA, tB, dimH, dimNB, sAb, sAh, sAn, sBb, sBh, sBn,
            sCb, sCh, sCn, sRb, epi);
    };
    auto ew = [&](long n) { return dim3((unsigned)((n + 255) / 256)); };

    // ---- init running memory state from inputs (deterministic, d2d) ----
    hipMemcpyAsync(W_new, W_prev, BDD * sizeof(float), hipMemcpyDeviceToDevice, stream);
    hipMemcpyAsync(m_new, m_prev, BDD * sizeof(float), hipMemcpyDeviceToDevice, stream);

    // =================== memory path ===================
    layernorm_kernel<<<Bb * S, 256, 0, stream>>>(x, ln1_g, ln1_b, xn, D);
    gemm(xn, mem_Wq, qm, nullptr, nullptr, 1.f, 0.f, Bb * S, D, D, D, D, D, 0,
         0, 0, 1, 1, 1, 0,0,0, 0,0,0, 0,0,0, 0, EPI_NONE);
    gemm(xn, mem_Wk, km, nullptr, nullptr, 1.f, 0.f, Bb * S, D, D, D, D, D, 0,
         0, 0, 1, 1, 1, 0,0,0, 0,0,0, 0,0,0, 0, EPI_NONE);
    gemm(xn, mem_Wv, vm, nullptr, nullptr, 1.f, 0.f, Bb * S, D, D, D, D, D, 0,
         0, 0, 1, 1, 1, 0,0,0, 0,0,0, 0,0,0, 0, EPI_NONE);

    // err_full = k @ W_prev - v   (batched over b)
    gemm(km, W_prev, errbuf, nullptr, vm, 1.f, -1.f, S, D, D, D, D, D, D,
         0, 0, Bb, 1, 1, (long)S * D, 0, 0, DD, 0, 0, (long)S * D, 0, 0,
         (long)S * D, EPI_NONE);
    // grad_full = k^T @ err / S
    gemm(km, errbuf, gradb, nullptr, nullptr, 1.f / S, 0.f, D, D, S, D, D, D, 0,
         1, 0, Bb, 1, 1, (long)S * D, 0, 0, (long)S * D, 0, 0, DD, 0, 0,
         0, EPI_NONE);
    retgrad_kernel<<<ew(BDD), 256, 0, stream>>>(W_prev, m_prev, gradb,
                                                mem_lr, ret_l, ret_g, retgr, BDD);

    // chunked recurrent update (NC sequential steps; state in W_new/m_new)
    for (int c = 0; c < NC; c++) {
        const float* kc = km + (long)c * CH * D;
        const float* vc = vm + (long)c * CH * D;
        const float* qc = qm + (long)c * CH * D;
        // err_c = kc @ Wm - vc
        gemm(kc, W_new, errbuf, nullptr, vc, 1.f, -1.f, CH, D, D, D, D, D, D,
             0, 0, Bb, 1, 1, (long)S * D, 0, 0, DD, 0, 0, (long)CH * D, 0, 0,
             (long)S * D, EPI_NONE);
        // g_c = kc^T @ err_c / CH
        gemm(kc, errbuf, gradb, nullptr, nullptr, 1.f / CH, 0.f, D, D, CH,
             D, D, D, 0, 1, 0, Bb, 1, 1, (long)S * D, 0, 0, (long)CH * D, 0, 0,
             DD, 0, 0, 0, EPI_NONE);
        memstep_kernel<<<ew(BDD), 256, 0, stream>>>(W_new, m_new, gradb, retgr,
                                                    mem_lr, BDD);
        // out_c = qc @ Wm
        gemm(qc, W_new, mem_out + (long)c * CH * D, nullptr, nullptr, 1.f, 0.f,
             CH, D, D, D, D, D, 0, 0, 0, Bb, 1, 1, (long)S * D, 0, 0, DD, 0, 0,
             (long)S * D, 0, 0, 0, EPI_NONE);
    }

    // =================== sliding-window attention ===================
    layernorm_kernel<<<Bb * S, 256, 0, stream>>>(x, ln2_g, ln2_b, xn, D);
    // zero the WIN-row pads in front of each batch of k/v
    for (int b = 0; b < Bb; b++) {
        hipMemsetAsync(ka_pad + (long)b * SP, 0, (long)WIN * D * sizeof(float), stream);
        hipMemsetAsync(va_pad + (long)b * SP, 0, (long)WIN * D * sizeof(float), stream);
    }
    gemm(xn, aWq, qa, abq, nullptr, 1.f, 0.f, Bb * S, D, D, D, D, D, 0,
         0, 0, 1, 1, 1, 0,0,0, 0,0,0, 0,0,0, 0, EPI_NONE);
    gemm(xn, aWk, ka_pad + (long)WIN * D, abk, nullptr, 1.f, 0.f, S, D, D,
         D, D, D, 0, 0, 0, Bb, 1, 1, (long)S * D, 0, 0, 0, 0, 0, SP, 0, 0,
         0, EPI_NONE);
    gemm(xn, aWv, va_pad + (long)WIN * D, abv, nullptr, 1.f, 0.f, S, D, D,
         D, D, D, 0, 0, 0, Bb, 1, 1, (long)S * D, 0, 0, 0, 0, 0, SP, 0, 0,
         0, EPI_NONE);

    // scores[z=(b*H+h)*NB+nb][512][1024] = (Q . K^T)/8
    gemm(qa, ka_pad, scores, nullptr, nullptr, 0.125f, 0.f,
         WIN, 2 * WIN, HD, D, D, 2 * WIN, 0, 0, 1,
         Bb * H * NB, H, NB,
         (long)S * D,  HD, (long)WIN * D,          // A = qa
         SP,           HD, (long)WIN * D,          // B = ka_pad (transB)
         SCB, (long)NB * WIN * 2 * WIN, (long)WIN * 2 * WIN,  // C = scores
         0, EPI_NONE);
    softmax_band<<<Bb * H * NB * WIN, 256, 0, stream>>>(scores, NB);
    // attn = P @ V
    gemm(scores, va_pad, attn, nullptr, nullptr, 1.f, 0.f,
         WIN, HD, 2 * WIN, 2 * WIN, D, D, 0, 0, 0,
         Bb * H * NB, H, NB,
         SCB, (long)NB * WIN * 2 * WIN, (long)WIN * 2 * WIN,  // A = probs
         SP,           HD, (long)WIN * D,                      // B = va_pad
         (long)S * D,  HD, (long)WIN * D,                      // C = attn
         0, EPI_NONE);
    gemm(attn, aWo, attn_o, abo, nullptr, 1.f, 0.f, Bb * S, D, D, D, D, D, 0,
         0, 0, 1, 1, 1, 0,0,0, 0,0,0, 0,0,0, 0, EPI_NONE);

    // =================== gated fusion + FFN ===================
    // gate = sigmoid(mem_out @ Wg_top + attn_o @ Wg_bot + b)
    gemm(mem_out, gateW, gate, gateb, nullptr, 1.f, 0.f, Bb * S, D, D,
         D, D, D, 0, 0, 0, 1, 1, 1, 0,0,0, 0,0,0, 0,0,0, 0, EPI_NONE);
    gemm(attn_o, gateW + DD, gate, nullptr, gate, 1.f, 1.f, Bb * S, D, D,
         D, D, D, D, 0, 0, 1, 1, 1, 0,0,0, 0,0,0, 0,0,0, 0, EPI_SIGMOID);
    fuse_kernel<<<ew(BSD), 256, 0, stream>>>(x, gate, mem_out, attn_o, x2, BSD);

    layernorm_kernel<<<Bb * S, 256, 0, stream>>>(x2, ln3_g, ln3_b, xn, D);
    gemm(xn, fW1, hbuf, fb1, nullptr, 1.f, 0.f, Bb * S, 4 * D, D,
         D, 4 * D, 4 * D, 0, 0, 0, 1, 1, 1, 0,0,0, 0,0,0, 0,0,0, 0, EPI_GELU);
    gemm(hbuf, fW2, out_x, fb2, x2, 1.f, 1.f, Bb * S, D, 4 * D,
         4 * D, D, D, D, 0, 0, 1, 1, 1, 0,0,0, 0,0,0, 0,0,0, 0, EPI_NONE);
}
